// FNOBlock_15066745274431
// MI455X (gfx1250) — matrix operand order repaired and checked by reading, not compiled, not run on hardware
//
#include <hip/hip_runtime.h>
#include <hip/hip_bf16.h>
#include <math.h>

// ---------------------------------------------------------------------------
// FNO block for MI455X (gfx1250, wave32, WMMA).
// All stages are bf16 WMMA GEMMs (f32 accumulate). Truncated DFTs (only 24x24
// modes survive) are dense matmuls against baked twiddle tables. LDS tiles are
// stored in WMMA *fragment order* so every lane loads its v16bf operands as
// single contiguous 32B LDS reads; all B tables are baked N-major so staging
// is pure b128 copies.
// ---------------------------------------------------------------------------

typedef __attribute__((ext_vector_type(16))) __bf16 v16bf;
typedef __attribute__((ext_vector_type(4)))  __bf16 v4bf;
typedef __attribute__((ext_vector_type(8)))  float  v8f;

#define BB 16
#define CC 64
#define HH 128
#define WW 128
#define HID 32
#define PI2 6.28318530717958647692f

__device__ inline v8f vzero8() {
  v8f v;
#pragma unroll
  for (int i = 0; i < 8; ++i) v[i] = 0.0f;
  return v;
}

__device__ inline float gelu_exact(float x) {
  return 0.5f * x * (1.0f + erff(x * 0.70710678118654752f));
}

// Fragment-order index for the 16-bit A matrix (16x32), wave32:
// lane = 16*(k>>4) + m holds element i = k&15.
__device__ inline int a_frag_idx(int m, int k) {
  return (((k >> 4) << 4) + m) * 16 + (k & 15);
}

// One 16x16 tile, one K=32 chunk. Afrag: 512 bf16 in fragment order.
// Bt: N-major [n][ldk] bf16, caller pre-offsets by ncol0*ldk + kchunk.
__device__ inline v8f bf16_mma(const __bf16* Afrag, const __bf16* Bt, int ldk,
                               v8f acc, int lane) {
  const int half = lane >> 4, m = lane & 15;
  const v16bf a = *(const v16bf*)(Afrag + lane * 16);
  const v16bf b = *(const v16bf*)(Bt + m * ldk + half * 16);
  return __builtin_amdgcn_wmma_f32_16x16x32_bf16(false, a, false, b,
                                                 (short)0, acc, false, false);
}

__device__ inline v4bf cvt4(float4 f) {
  v4bf t;
  t[0] = (__bf16)f.x; t[1] = (__bf16)f.y; t[2] = (__bf16)f.z; t[3] = (__bf16)f.w;
  return t;
}

// ---------------------------------------------------------------------------
// Table bake (all B matrices N-major: Bt[n][k] = B[k][n]).
// ---------------------------------------------------------------------------
__global__ __launch_bounds__(256) void k_tables(
    const float* __restrict__ wskip, const float* __restrict__ fc1w,
    const float* __restrict__ fc2w,
    __bf16* __restrict__ EWt,   // [64 j][128 w]   fwd rDFT along W
    __bf16* __restrict__ EHt,   // [64 j][256 r]   fwd DFT along H (stacked K)
    __bf16* __restrict__ EiHt,  // [256 n][64 k]   inv DFT along H
    __bf16* __restrict__ EiWt,  // [128 w][64 k]   Hermitian inv rDFT along W
    __bf16* __restrict__ Wsk,   // [64 o][64 c]    = bf16(w_skip)
    __bf16* __restrict__ F1b,   // [32 k][64 c]    = bf16(fc1_w)
    __bf16* __restrict__ F2b) { // [64 o][32 k]    = bf16(fc2_w)
  const int t = blockIdx.x * blockDim.x + threadIdx.x; // 0..16383
  if (t < 64 * 128) { // EWt[j][w]
    const int j = t >> 7, w = t & 127;
    float v = 0.0f;
    if (j < 24)      v =  cosf(PI2 * (float)(w * j) / 128.0f);
    else if (j < 48) v = -sinf(PI2 * (float)(w * (j - 24)) / 128.0f);
    EWt[t] = (__bf16)v;
  }
  if (t < 64 * 256) { // EHt[j][r]: r<128 -> T1r row h=r, r>=128 -> T1i row
    const int j = t >> 8, r = t & 255;
    const int h = r & 127; const bool im_row = r >= 128;
    float v = 0.0f;
    if (j < 24) {
      const float th = PI2 * (float)(h * j) / 128.0f;
      v = im_row ? sinf(th) : cosf(th);
    } else if (j < 48) {
      const float th = PI2 * (float)(h * (j - 24)) / 128.0f;
      v = im_row ? cosf(th) : -sinf(th);
    }
    EHt[t] = (__bf16)v;
  }
  if (t < 256 * 64) { // EiHt[n][k]: n<128 -> gr col h=n, n>=128 -> gi col
    const int n = t >> 6, k = t & 63;
    const int h = n & 127; const bool gi_col = n >= 128;
    float v = 0.0f;
    if (k < 24) {
      const float th = PI2 * (float)(k * h) / 128.0f;
      v = (gi_col ? sinf(th) : cosf(th)) * (1.0f / 128.0f);
    } else if (k < 48) {
      const float th = PI2 * (float)((k - 24) * h) / 128.0f;
      v = (gi_col ? cosf(th) : -sinf(th)) * (1.0f / 128.0f);
    }
    EiHt[t] = (__bf16)v;
  }
  if (t < 128 * 64) { // EiWt[w][k], imag of ky=0 dropped (k==24 row -> 0)
    const int w = t >> 6, k = t & 63;
    float v = 0.0f;
    if (k == 0)      v = 1.0f / 128.0f;
    else if (k < 24) v =  2.0f * cosf(PI2 * (float)(k * w) / 128.0f) / 128.0f;
    else if (k < 48) {
      const int ky = k - 24;
      if (ky > 0) v = -2.0f * sinf(PI2 * (float)(ky * w) / 128.0f) / 128.0f;
    }
    EiWt[t] = (__bf16)v;
  }
  if (t < CC * CC)  Wsk[t] = (__bf16)wskip[t];  // already [o][c]
  if (t < HID * CC) F1b[t] = (__bf16)fc1w[t];   // already [k][c]
  if (t < CC * HID) F2b[t] = (__bf16)fc2w[t];   // already [o][k]
}

// Mode-mix weights, N-major: Wmt[mode][j][r], j = out col ([Gr|Gi] over o),
// r = stacked K ([Fr rows c | Fi rows c]).
__global__ __launch_bounds__(256) void k_wmode(const float* __restrict__ wr,
                                               const float* __restrict__ wi,
                                               __bf16* __restrict__ Wmt) {
  const int t = blockIdx.x * blockDim.x + threadIdx.x;
  if (t >= 576 * 128 * 128) return;
  const int mode = t >> 14;
  const int j = (t >> 7) & 127, r = t & 127;
  const int o = j & 63, c = r & 63;
  const bool im_col = j >= 64, im_row = r >= 64;
  const int widx = (mode * 64 + c) * 64 + o;
  float v;
  if (!im_row) v = im_col ? wi[widx] : wr[widx];
  else         v = im_col ? wr[widx] : -wi[widx];
  Wmt[t] = (__bf16)v;
}

// Vectorized B-chunk staging: copy Bt rows [0..N) x [kc*32, kc*32+32) into
// LDS chunk (N-major, ldk 32), 16B per thread-iteration.
#define STAGE_B(BsLds, BTglob, N, KTOT, kc)                                   \
  for (int e = threadIdx.x; e < (N) * 4; e += 128) {                          \
    const int n_ = e >> 2, q_ = e & 3;                                        \
    *(uint4*)((BsLds) + n_ * 32 + q_ * 8) =                                   \
        *(const uint4*)((BTglob) + (long long)n_ * (KTOT) + (kc) * 32 + q_ * 8); \
  }

// ---------------------------------------------------------------------------
// Stage A: truncated rDFT along W. rows=(b,c,h)=131072 contiguous len-128.
// Output T1t[bc][j(64)][h(128)] (j: 0..23 Re, 24..47 Im, 48..63 zero pad).
// ---------------------------------------------------------------------------
__global__ __launch_bounds__(128) void k_dftw(const float* __restrict__ x,
                                              const __bf16* __restrict__ EWt,
                                              float* __restrict__ T1t) {
  __shared__ alignas(32) __bf16 As[512];
  __shared__ alignas(32) __bf16 Bs[64 * 32];
  const int t = threadIdx.x, wave = t >> 5, lane = t & 31;
  const int row0 = blockIdx.x * 16;
  const int r = t >> 3, ks = (t & 7) * 4;
  v8f acc = vzero8();
  for (int kc = 0; kc < 4; ++kc) {
    const float4 f4 = *(const float4*)(x + (long long)(row0 + r) * 128 + kc * 32 + ks);
    *(v4bf*)(As + a_frag_idx(r, ks)) = cvt4(f4);
    if (kc < 3) __builtin_prefetch(x + (long long)(row0 + r) * 128 + (kc + 1) * 32 + ks, 0, 0);
    STAGE_B(Bs, EWt, 64, 128, kc);
    __syncthreads();
    acc = bf16_mma(As, Bs + (wave * 16) * 32, 32, acc, lane);
    __syncthreads();
  }
  const int half = lane >> 4, m = lane & 15;
  const int j = wave * 16 + m;
  const long long base =
      ((long long)(row0 >> 7) * 64 + j) * 128 + (row0 & 127) + 8 * half;
  *(float4*)(T1t + base)     = make_float4(acc[0], acc[1], acc[2], acc[3]);
  *(float4*)(T1t + base + 4) = make_float4(acc[4], acc[5], acc[6], acc[7]);
}

// ---------------------------------------------------------------------------
// Stage B: DFT along H (stacked K=256). rows=(bc,ky)=24576.
// Output F2[kx][im][ky][b][c] (contiguous in c for stage C).
// ---------------------------------------------------------------------------
__global__ __launch_bounds__(128) void k_dfth(const float* __restrict__ T1t,
                                              const __bf16* __restrict__ EHt,
                                              float* __restrict__ F2) {
  __shared__ alignas(32) __bf16 As[512];
  __shared__ alignas(32) __bf16 Bs[64 * 32];
  const int t = threadIdx.x, wave = t >> 5, lane = t & 31;
  const int row0 = blockIdx.x * 16;
  const int r = t >> 3, ks = (t & 7) * 4;
  const int rid = row0 + r;
  const int bc = rid / 24, ky = rid % 24;
  v8f acc = vzero8();
  for (int kc = 0; kc < 8; ++kc) {
    const int k = kc * 32 + ks;
    const float* src = (k < 128)
        ? T1t + ((long long)bc * 64 + ky) * 128 + k
        : T1t + ((long long)bc * 64 + 24 + ky) * 128 + (k - 128);
    *(v4bf*)(As + a_frag_idx(r, ks)) = cvt4(*(const float4*)src);
    STAGE_B(Bs, EHt, 64, 256, kc);
    __syncthreads();
    acc = bf16_mma(As, Bs + (wave * 16) * 32, 32, acc, lane);
    __syncthreads();
  }
  const int half = lane >> 4, m = lane & 15;
  const int j = wave * 16 + m;
  if (j < 48) {
    const int im = j >= 24 ? 1 : 0;
    const int kx = j - 24 * im;
#pragma unroll
    for (int v = 0; v < 8; ++v) {
      const int rid2 = row0 + v + 8 * half;
      const int bc2 = rid2 / 24, ky2 = rid2 % 24;
      F2[((((long long)kx * 2 + im) * 24 + ky2) * 16 + (bc2 >> 6)) * 64 + (bc2 & 63)] = acc[v];
    }
  }
}

// ---------------------------------------------------------------------------
// Stage C: per-mode complex channel mix (stacked K=128). One block per mode.
// Output G2[ky][b][o][j(64)] with j = kx + 24*im (contiguous K for stage D).
// ---------------------------------------------------------------------------
__global__ __launch_bounds__(128) void k_modemix(const float* __restrict__ F2,
                                                 const __bf16* __restrict__ Wmt,
                                                 float* __restrict__ G2) {
  __shared__ alignas(32) __bf16 As[512];
  __shared__ alignas(32) __bf16 Bs[128 * 32];
  const int t = threadIdx.x, wave = t >> 5, lane = t & 31;
  const int mode = blockIdx.x;
  const int kx = mode / 24, ky = mode % 24;
  const int r = t >> 3, ks = (t & 7) * 4; // r = batch row
  v8f acc0 = vzero8(), acc1 = vzero8();
  for (int kc = 0; kc < 4; ++kc) {
    const int k = kc * 32 + ks;
    const int im = k >= 64 ? 1 : 0;
    const float* src =
        F2 + ((((long long)kx * 2 + im) * 24 + ky) * 16 + r) * 64 + (k - 64 * im);
    *(v4bf*)(As + a_frag_idx(r, ks)) = cvt4(*(const float4*)src);
    STAGE_B(Bs, Wmt + (long long)mode * 16384, 128, 128, kc);
    __syncthreads();
    acc0 = bf16_mma(As, Bs + (wave * 16) * 32, 32, acc0, lane);
    acc1 = bf16_mma(As, Bs + (64 + wave * 16) * 32, 32, acc1, lane);
    __syncthreads();
  }
  const int half = lane >> 4, m = lane & 15;
  const int o = wave * 16 + m;
#pragma unroll
  for (int v = 0; v < 8; ++v) {
    const int b = v + 8 * half;
    const long long base = (((long long)ky * 16 + b) * 64 + o) * 64;
    G2[base + kx]      = acc0[v];
    G2[base + 24 + kx] = acc1[v];
  }
}

// ---------------------------------------------------------------------------
// Stage D: inverse DFT along H (K=48 pad 64). rows=(b,o,ky)=24576, N=256.
// Output g2[bo][h][j(64)] with j = ky + 24*im (contiguous K for stage E).
// ---------------------------------------------------------------------------
__global__ __launch_bounds__(128) void k_idfth(const float* __restrict__ G2,
                                               const __bf16* __restrict__ EiHt,
                                               float* __restrict__ g2) {
  __shared__ alignas(32) __bf16 As[512];
  __shared__ alignas(32) __bf16 Bs[256 * 32];
  const int t = threadIdx.x, wave = t >> 5, lane = t & 31;
  const int row0 = blockIdx.x * 16;
  const int r = t >> 3, ks = (t & 7) * 4;
  const int rid = row0 + r;
  const int ky = rid % 24, bo = rid / 24;
  const long long abase = (((long long)ky * 16 + (bo >> 6)) * 64 + (bo & 63)) * 64;
  v8f acc[4];
#pragma unroll
  for (int i = 0; i < 4; ++i) acc[i] = vzero8();
  for (int kc = 0; kc < 2; ++kc) {
    const int k = kc * 32 + ks;
    v4bf t4;
    if (k < 48) t4 = cvt4(*(const float4*)(G2 + abase + k));
    else { t4[0] = (__bf16)0.0f; t4[1] = (__bf16)0.0f; t4[2] = (__bf16)0.0f; t4[3] = (__bf16)0.0f; }
    *(v4bf*)(As + a_frag_idx(r, ks)) = t4;
    STAGE_B(Bs, EiHt, 256, 64, kc);
    __syncthreads();
#pragma unroll
    for (int nt = 0; nt < 4; ++nt)
      acc[nt] = bf16_mma(As, Bs + ((wave * 4 + nt) * 16) * 32, 32, acc[nt], lane);
    __syncthreads();
  }
  const int half = lane >> 4, m = lane & 15;
#pragma unroll
  for (int nt = 0; nt < 4; ++nt) {
    const int n = (wave * 4 + nt) * 16 + m;
    const int h = n & 127, im = n >= 128 ? 1 : 0;
#pragma unroll
    for (int v = 0; v < 8; ++v) {
      const int rid2 = row0 + v + 8 * half;
      const int ky2 = rid2 % 24, bo2 = rid2 / 24;
      g2[((long long)bo2 * 128 + h) * 64 + ky2 + 24 * im] = acc[nt][v];
    }
  }
}

// ---------------------------------------------------------------------------
// Stage E: Hermitian inverse rDFT along W (K=48 pad 64). rows=(bo,h), N=128.
// Writes spectral output into h1 buffer (pre-skip).
// ---------------------------------------------------------------------------
__global__ __launch_bounds__(128) void k_idftw(const float* __restrict__ g2,
                                               const __bf16* __restrict__ EiWt,
                                               float* __restrict__ h1) {
  __shared__ alignas(32) __bf16 As[512];
  __shared__ alignas(32) __bf16 Bs[128 * 32];
  const int t = threadIdx.x, wave = t >> 5, lane = t & 31;
  const int row0 = blockIdx.x * 16;
  const int r = t >> 3, ks = (t & 7) * 4;
  const long long abase = (long long)(row0 + r) * 64;
  v8f acc0 = vzero8(), acc1 = vzero8();
  for (int kc = 0; kc < 2; ++kc) {
    const int k = kc * 32 + ks;
    v4bf t4;
    if (k < 48) t4 = cvt4(*(const float4*)(g2 + abase + k));
    else { t4[0] = (__bf16)0.0f; t4[1] = (__bf16)0.0f; t4[2] = (__bf16)0.0f; t4[3] = (__bf16)0.0f; }
    *(v4bf*)(As + a_frag_idx(r, ks)) = t4;
    STAGE_B(Bs, EiWt, 128, 64, kc);
    __syncthreads();
    acc0 = bf16_mma(As, Bs + (wave * 16) * 32, 32, acc0, lane);
    acc1 = bf16_mma(As, Bs + (64 + wave * 16) * 32, 32, acc1, lane);
    __syncthreads();
  }
  const int half = lane >> 4, m = lane & 15;
#pragma unroll
  for (int v = 0; v < 8; ++v) {
    const long long rid2 = row0 + v + 8 * half; // (b*64+o)*128 + h
    h1[rid2 * 128 + wave * 16 + m]      = acc0[v];
    h1[rid2 * 128 + 64 + wave * 16 + m] = acc1[v];
  }
}

// ---------------------------------------------------------------------------
// Stage F: 1x1-conv skip + GELU fused in place: h1 = gelu(spec + x*Wskip^T).
// ---------------------------------------------------------------------------
__global__ __launch_bounds__(128) void k_skip(const float* __restrict__ x,
                                              const __bf16* __restrict__ Wsk,
                                              float* __restrict__ h1) {
  __shared__ alignas(32) __bf16 As[512];
  __shared__ alignas(32) __bf16 Bs[64 * 32];
  const int t = threadIdx.x, wave = t >> 5, lane = t & 31;
  const int b = blockIdx.x >> 10;
  const int pix0 = (blockIdx.x & 1023) * 16;
  const float* xb = x + (long long)b * CC * (HH * WW);
  v8f acc = vzero8();
  for (int kc = 0; kc < 2; ++kc) {
    for (int e = t; e < 512; e += 128) {
      const int p = e & 15, cchunk = e >> 4;
      As[a_frag_idx(p, cchunk)] =
          (__bf16)xb[(long long)(kc * 32 + cchunk) * (HH * WW) + pix0 + p];
    }
    STAGE_B(Bs, Wsk, 64, 64, kc);
    __syncthreads();
    acc = bf16_mma(As, Bs + (wave * 16) * 32, 32, acc, lane);
    __syncthreads();
  }
  const int half = lane >> 4, m = lane & 15;
  const int o = wave * 16 + m;
#pragma unroll
  for (int v = 0; v < 8; ++v) {
    const long long idx = ((long long)b * CC + o) * (HH * WW) + pix0 + v + 8 * half;
    h1[idx] = gelu_exact(acc[v] + h1[idx]);
  }
}

// ---------------------------------------------------------------------------
// Stage G+H fused: per-pixel channel MLP + gated residual.
// ---------------------------------------------------------------------------
__global__ __launch_bounds__(128) void k_mlp(const float* __restrict__ h1,
                                             const __bf16* __restrict__ F1b,
                                             const __bf16* __restrict__ F2b,
                                             const float* __restrict__ b1,
                                             const float* __restrict__ b2,
                                             const float* __restrict__ gate,
                                             float* __restrict__ out) {
  __shared__ alignas(32) __bf16 Hs[4 * 512]; // [mt][kc] fragment tiles
  __shared__ alignas(32) __bf16 Zs[2 * 512]; // [mt] fragment tiles (K=32)
  __shared__ alignas(32) __bf16 F1s[32 * 64];
  __shared__ alignas(32) __bf16 F2s[64 * 32];
  const int t = threadIdx.x, wave = t >> 5, lane = t & 31;
  const int b = blockIdx.x >> 9;
  const int pix0 = (blockIdx.x & 511) * 32;
  const float* hb = h1 + (long long)b * CC * (HH * WW);
  for (int e = t; e < 2048; e += 128) {
    const int p = e & 31, c = e >> 5;
    Hs[((p >> 4) * 2 + (c >> 5)) * 512 + a_frag_idx(p & 15, c & 31)] =
        (__bf16)hb[(long long)c * (HH * WW) + pix0 + p];
  }
  for (int e = t; e < 256; e += 128) {
    *(uint4*)(F1s + e * 8) = *(const uint4*)(F1b + e * 8);
    *(uint4*)(F2s + e * 8) = *(const uint4*)(F2b + e * 8);
  }
  __syncthreads();
  const int half = lane >> 4, m = lane & 15;
  // GEMM1: z = gelu(h1 * fc1^T + b1), K=64, result into Zs fragments
  {
    const int mt = wave >> 1, nt = wave & 1;
    v8f acc = vzero8();
#pragma unroll
    for (int kc = 0; kc < 2; ++kc)
      acc = bf16_mma(Hs + (mt * 2 + kc) * 512, F1s + (nt * 16) * 64 + kc * 32,
                     64, acc, lane);
    const int kcol = nt * 16 + m;
    const float bias = b1[kcol];
#pragma unroll
    for (int v = 0; v < 8; ++v)
      Zs[mt * 512 + a_frag_idx(v + 8 * half, kcol)] =
          (__bf16)gelu_exact(acc[v] + bias);
  }
  __syncthreads();
  // GEMM2: out = z * fc2^T + b2 + gate*h1, K=32
  {
    const int mt = wave >> 1;
#pragma unroll
    for (int q = 0; q < 2; ++q) {
      const int nt = (wave & 1) + q * 2;
      v8f acc = vzero8();
      acc = bf16_mma(Zs + mt * 512, F2s + (nt * 16) * 32, 32, acc, lane);
      const int o = nt * 16 + m;
      const float bias = b2[o], gg = gate[o];
#pragma unroll
      for (int v = 0; v < 8; ++v) {
        const int p = mt * 16 + v + 8 * half;
        const long long idx = ((long long)b * CC + o) * (HH * WW) + pix0 + p;
        out[idx] = acc[v] + bias + gg * h1[idx];
      }
    }
  }
}

// ---------------------------------------------------------------------------
extern "C" void kernel_launch(void* const* d_in, const int* in_sizes, int n_in,
                              void* d_out, int out_size, void* d_ws, size_t ws_size,
                              hipStream_t stream) {
  (void)in_sizes; (void)n_in; (void)out_size; (void)ws_size;
  const float* x     = (const float*)d_in[0];
  const float* wr    = (const float*)d_in[1];
  const float* wi    = (const float*)d_in[2];
  const float* wskip = (const float*)d_in[3];
  const float* fc1w  = (const float*)d_in[4];
  const float* fc1b  = (const float*)d_in[5];
  const float* fc2w  = (const float*)d_in[6];
  const float* fc2b  = (const float*)d_in[7];
  const float* gate  = (const float*)d_in[8];
  float* out = (float*)d_out;

  char* ws = (char*)d_ws;
  size_t off = 0;
  auto take = [&](size_t bytes) -> char* {
    char* p = ws + off;
    off = (off + bytes + 255) & ~(size_t)255;
    return p;
  };
  __bf16* EWt  = (__bf16*)take(64 * 128 * 2);
  __bf16* EHt  = (__bf16*)take(64 * 256 * 2);
  __bf16* EiHt = (__bf16*)take(256 * 64 * 2);
  __bf16* EiWt = (__bf16*)take(128 * 64 * 2);
  __bf16* Wsk  = (__bf16*)take(64 * 64 * 2);
  __bf16* F1b  = (__bf16*)take(32 * 64 * 2);
  __bf16* F2b  = (__bf16*)take(64 * 32 * 2);
  __bf16* Wmt  = (__bf16*)take((size_t)576 * 128 * 128 * 2);        // 18.9 MB
  float*  F2   = (float*)take((size_t)24 * 2 * 24 * 16 * 64 * 4);   //  4.7 MB
  float*  G2   = (float*)take((size_t)24 * 16 * 64 * 64 * 4);       //  6.3 MB
  float*  T1t  = (float*)take((size_t)1024 * 64 * 128 * 4);         // 33.6 MB
  float*  g2   = T1t; // T1t dead after stage B; reuse for iDFT-H output
  float*  h1   = (float*)take((size_t)BB * CC * HH * WW * 4);       // 67.1 MB

  k_tables<<<64, 256, 0, stream>>>(wskip, fc1w, fc2w, EWt, EHt, EiHt, EiWt,
                                   Wsk, F1b, F2b);
  k_wmode<<<36864, 256, 0, stream>>>(wr, wi, Wmt);
  k_dftw<<<(BB * CC * HH) / 16, 128, 0, stream>>>(x, EWt, T1t);
  k_dfth<<<(BB * CC * 24) / 16, 128, 0, stream>>>(T1t, EHt, F2);
  k_modemix<<<576, 128, 0, stream>>>(F2, Wmt, G2);
  k_idfth<<<(BB * CC * 24) / 16, 128, 0, stream>>>(G2, EiHt, g2);
  k_idftw<<<(BB * CC * HH) / 16, 128, 0, stream>>>(g2, EiWt, h1);
  k_skip<<<BB * (HH * WW / 16), 128, 0, stream>>>(x, Wsk, h1);
  k_mlp<<<BB * (HH * WW / 32), 128, 0, stream>>>(h1, F1b, F2b, fc1b, fc2b,
                                                 gate, out);
}
